// SimpleRNN_18597208391944
// MI455X (gfx1250) — compile-verified, matrix operand order checked by reading, not run on hardware
//
#include <hip/hip_runtime.h>
#include <math.h>

// SimpleRNN on MI455X (gfx1250, wave32, WMMA).
#define Bb 64
#define Tt 512
#define Ii 256
#define Hh 1024
#define Oo 256

typedef __attribute__((ext_vector_type(16))) __bf16 v16bf;
typedef __attribute__((ext_vector_type(8)))  __bf16 v8bf;
typedef __attribute__((ext_vector_type(8)))  float  v8f;
typedef __attribute__((ext_vector_type(4)))  float  v4f;

static __device__ __forceinline__ v16bf cat16(v8bf lo, v8bf hi) {
    v16bf r;
#pragma unroll
    for (int i = 0; i < 8; ++i) { r[i] = lo[i]; r[i + 8] = hi[i]; }
    return r;
}

// ---------------------------------------------------------------------------
// Pack src[N][K] (row-major f32) into the CDNA5 WMMA bf16 B-operand layout
// for 16x16x32 tiles: dst[((kt*NT + nt)*32 + lane)*16 + j]
//   lane: n_local = lane&15, hi = lane>>4
//   j   : k_local = (j&7) + (j>=8 ? 16 : 0) + hi*8   (ISA 05_wmma 16-bit layout)
// Each lane's 16 bf16 (32B) are contiguous -> two b128 loads in the GEMM.
// ---------------------------------------------------------------------------
__global__ void pack_wmma_b(const float* __restrict__ src, __bf16* __restrict__ dst,
                            int K, int N) {
    int idx = blockIdx.x * blockDim.x + threadIdx.x;
    int total = (K >> 5) * (N >> 4) * 32 * 16;
    if (idx >= total) return;
    int j  = idx & 15;
    int L  = (idx >> 4) & 31;
    int nt = (idx >> 9) % (N >> 4);
    int kt = (idx >> 9) / (N >> 4);
    int hi = L >> 4;
    int k  = kt * 32 + (j & 7) + ((j >= 8) ? 16 : 0) + hi * 8;
    int n  = nt * 16 + (L & 15);
    dst[idx] = (__bf16)src[(size_t)n * K + k];
}

__global__ void zero_u32(unsigned* __restrict__ p, int n) {
    int i = blockIdx.x * blockDim.x + threadIdx.x;
    if (i < n) p[i] = 0u;
}

// ---------------------------------------------------------------------------
// xp[t*B + b][h] = x[b][t][:] . W_ih[h][:] + b_ih[h] + b_hh[h]   (f32 out)
// One wave per 16x16 tile, K=256 (8 bf16 WMMAs), fully parallel (131072 waves).
// ---------------------------------------------------------------------------
__global__ void xp_gemm(const float* __restrict__ x, const __bf16* __restrict__ wih_pk,
                        const float* __restrict__ b_ih, const float* __restrict__ b_hh,
                        float* __restrict__ xp) {
    int wave = blockIdx.x * (blockDim.x >> 5) + (threadIdx.x >> 5);
    int lane = threadIdx.x & 31;
    int nt = wave & (Hh / 16 - 1);   // 0..63
    int rm = wave >> 6;              // 0..2047
    int hi = lane >> 4;
    int ml = lane & 15;

    int r = rm * 16 + ml;            // output row = t*B + b
    int t = r >> 6, b = r & 63;
    const float* xrow = x + ((size_t)b * Tt + t) * Ii;

    v8f acc = {};
#pragma unroll
    for (int kt = 0; kt < Ii / 32; ++kt) {
        int k0 = kt * 32 + hi * 8;
        v4f c0 = *(const v4f*)(xrow + k0);
        v4f c1 = *(const v4f*)(xrow + k0 + 4);
        v4f c2 = *(const v4f*)(xrow + k0 + 16);
        v4f c3 = *(const v4f*)(xrow + k0 + 20);
        v16bf a;
#pragma unroll
        for (int i = 0; i < 4; ++i) {
            a[i]      = (__bf16)c0[i];
            a[i + 4]  = (__bf16)c1[i];
            a[i + 8]  = (__bf16)c2[i];
            a[i + 12] = (__bf16)c3[i];
        }
        const __bf16* bp = wih_pk + (((size_t)kt * (Hh / 16) + nt) * 32 + lane) * 16;
        v16bf bb = cat16(*(const v8bf*)bp, *(const v8bf*)(bp + 8));
        acc = __builtin_amdgcn_wmma_f32_16x16x32_bf16(false, a, false, bb,
                                                      (short)0, acc, false, false);
    }
    int col = nt * 16 + ml;
    float bias = b_ih[col] + b_hh[col];
#pragma unroll
    for (int vr = 0; vr < 8; ++vr) {
        int row = rm * 16 + hi * 8 + vr;       // C layout: M = vr + 8*hi
        xp[(size_t)row * Hh + col] = acc[vr] + bias;
    }
}

// ---------------------------------------------------------------------------
// Device-wide sense barrier (counter + generation word).
// ---------------------------------------------------------------------------
static __device__ __forceinline__ void grid_barrier(unsigned* __restrict__ sync, int nblocks) {
    __threadfence();
    __syncthreads();
    if (threadIdx.x == 0) {
        unsigned gen = __atomic_load_n(sync + 1, __ATOMIC_RELAXED);
        unsigned arrived = __atomic_fetch_add(sync, 1u, __ATOMIC_ACQ_REL);
        if (arrived == (unsigned)nblocks - 1u) {
            __atomic_store_n(sync, 0u, __ATOMIC_RELAXED);
            __threadfence();
            __atomic_store_n(sync + 1, gen + 1u, __ATOMIC_RELEASE);
        } else {
            while (__atomic_load_n(sync + 1, __ATOMIC_ACQUIRE) == gen)
                __builtin_amdgcn_s_sleep(1);
        }
    }
    __syncthreads();
}

// ---------------------------------------------------------------------------
// Persistent scan, split-K x4: 128 blocks x 8 waves = 1024 waves.
// Block owns 2 adjacent 16x16 h'-tiles (same bm -> A rows shared via WGP$);
// each tile is computed by 4 k-waves (8 WMMAs each, dual accumulators),
// partials reduced through LDS, k-wave 0 applies xp + tanh and stores bf16.
// h double-buffered; after 512 steps (even) the result is back in h0.
// ---------------------------------------------------------------------------
__global__ void rnn_scan(const __bf16* __restrict__ whh_pk,
                         const float* __restrict__ xp,
                         __bf16* __restrict__ h0, __bf16* __restrict__ h1,
                         unsigned* __restrict__ sync, int nblocks) {
    __shared__ float red[2][4][32][8];        // [tile][kwave][lane][vr], 8 KB

    int wv   = threadIdx.x >> 5;
    int lane = threadIdx.x & 31;
    int tl = wv >> 2;                         // 0..1  tile within block
    int kw = wv & 3;                          // 0..3  K-slice
    int tile = blockIdx.x * 2 + tl;           // 0..255
    int bm = tile >> 6;                       // 0..3
    int bn = tile & 63;                       // 0..63
    int hi = lane >> 4;
    int ml = lane & 15;
    int col = bn * 16 + ml;

    __bf16* cur = h0;
    __bf16* nxt = h1;

    for (int t = 0; t < Tt; ++t) {
        v8f acc0 = {}, acc1 = {};
        const __bf16* arow = cur + (size_t)(bm * 16 + ml) * Hh;
        int ktb = kw * 8;                     // this wave's K-slice: 8 of 32 kt
#pragma unroll
        for (int kk = 0; kk < 8; kk += 2) {
            int kt = ktb + kk;
            {   // even kt -> acc0
                int k0 = kt * 32 + hi * 8;
                v16bf a = cat16(*(const v8bf*)(arow + k0), *(const v8bf*)(arow + k0 + 16));
                const __bf16* bp = whh_pk + (((size_t)kt * (Hh / 16) + bn) * 32 + lane) * 16;
                __builtin_prefetch((const void*)(bp + (size_t)(Hh / 16) * 1024), 0, 1);
                v16bf bb = cat16(*(const v8bf*)bp, *(const v8bf*)(bp + 8));
                acc0 = __builtin_amdgcn_wmma_f32_16x16x32_bf16(false, a, false, bb,
                                                               (short)0, acc0, false, false);
            }
            {   // odd kt -> acc1 (independent chain)
                int k0 = (kt + 1) * 32 + hi * 8;
                v16bf a = cat16(*(const v8bf*)(arow + k0), *(const v8bf*)(arow + k0 + 16));
                const __bf16* bp = whh_pk + (((size_t)(kt + 1) * (Hh / 16) + bn) * 32 + lane) * 16;
                v16bf bb = cat16(*(const v8bf*)bp, *(const v8bf*)(bp + 8));
                acc1 = __builtin_amdgcn_wmma_f32_16x16x32_bf16(false, a, false, bb,
                                                               (short)0, acc1, false, false);
            }
        }
#pragma unroll
        for (int vr = 0; vr < 8; ++vr) red[tl][kw][lane][vr] = acc0[vr] + acc1[vr];
        __syncthreads();

        if (kw == 0) {
            const float* xpt = xp + ((size_t)t * Bb + bm * 16 + hi * 8) * Hh + col;
#pragma unroll
            for (int vr = 0; vr < 8; ++vr) {
                float s = red[tl][0][lane][vr] + red[tl][1][lane][vr]
                        + red[tl][2][lane][vr] + red[tl][3][lane][vr];
                float v = tanhf(s + xpt[(size_t)vr * Hh]);
                nxt[(size_t)(bm * 16 + hi * 8 + vr) * Hh + col] = (__bf16)v;
            }
        }
        grid_barrier(sync, nblocks);          // also protects LDS reuse (WAR)
        __bf16* tmp = cur; cur = nxt; nxt = tmp;
    }
}

// ---------------------------------------------------------------------------
// out[b][o] = h_last[b][:] . W_fc[o][:] + b_fc[o]  (f32), K = 1024,
// dual accumulators to break the WMMA chain.
// ---------------------------------------------------------------------------
__global__ void head_gemm(const __bf16* __restrict__ hfin, const __bf16* __restrict__ wfc_pk,
                          const float* __restrict__ b_fc, float* __restrict__ out) {
    int wv   = threadIdx.x >> 5;
    int lane = threadIdx.x & 31;
    int bm = blockIdx.x >> 1;                 // 0..3
    int bn = (blockIdx.x & 1) * 8 + wv;       // 0..15
    int hi = lane >> 4;
    int ml = lane & 15;

    const __bf16* arow = hfin + (size_t)(bm * 16 + ml) * Hh;
    v8f acc0 = {}, acc1 = {};
#pragma unroll 4
    for (int kt = 0; kt < Hh / 32; kt += 2) {
        {
            int k0 = kt * 32 + hi * 8;
            v16bf a = cat16(*(const v8bf*)(arow + k0), *(const v8bf*)(arow + k0 + 16));
            const __bf16* bp = wfc_pk + (((size_t)kt * (Oo / 16) + bn) * 32 + lane) * 16;
            v16bf bb = cat16(*(const v8bf*)bp, *(const v8bf*)(bp + 8));
            acc0 = __builtin_amdgcn_wmma_f32_16x16x32_bf16(false, a, false, bb,
                                                           (short)0, acc0, false, false);
        }
        {
            int k0 = (kt + 1) * 32 + hi * 8;
            v16bf a = cat16(*(const v8bf*)(arow + k0), *(const v8bf*)(arow + k0 + 16));
            const __bf16* bp = wfc_pk + (((size_t)(kt + 1) * (Oo / 16) + bn) * 32 + lane) * 16;
            v16bf bb = cat16(*(const v8bf*)bp, *(const v8bf*)(bp + 8));
            acc1 = __builtin_amdgcn_wmma_f32_16x16x32_bf16(false, a, false, bb,
                                                           (short)0, acc1, false, false);
        }
    }
    int col = bn * 16 + ml;
    float bias = b_fc[col];
#pragma unroll
    for (int vr = 0; vr < 8; ++vr)
        out[(size_t)(bm * 16 + hi * 8 + vr) * Oo + col] = acc0[vr] + acc1[vr] + bias;
}

// ---------------------------------------------------------------------------
// Workspace layout (~138 MB):
//   xp f32 [T*B][H] | wih_pk | whh_pk | wfc_pk | h0 | h1 | sync(2 u32)
// ---------------------------------------------------------------------------
extern "C" void kernel_launch(void* const* d_in, const int* in_sizes, int n_in,
                              void* d_out, int out_size, void* d_ws, size_t ws_size,
                              hipStream_t stream) {
    const float* x    = (const float*)d_in[0];
    const float* W_ih = (const float*)d_in[1];
    const float* W_hh = (const float*)d_in[2];
    const float* b_ih = (const float*)d_in[3];
    const float* b_hh = (const float*)d_in[4];
    const float* W_fc = (const float*)d_in[5];
    const float* b_fc = (const float*)d_in[6];

    char* ws = (char*)d_ws;
    size_t off = 0;
    float* xp = (float*)(ws + off);            off += (size_t)Tt * Bb * Hh * 4;
    __bf16* wih_pk = (__bf16*)(ws + off);      off += (size_t)(Ii/32) * (Hh/16) * 512 * 2;
    __bf16* whh_pk = (__bf16*)(ws + off);      off += (size_t)(Hh/32) * (Hh/16) * 512 * 2;
    __bf16* wfc_pk = (__bf16*)(ws + off);      off += (size_t)(Hh/32) * (Oo/16) * 512 * 2;
    __bf16* h0 = (__bf16*)(ws + off);          off += (size_t)Bb * Hh * 2;
    __bf16* h1 = (__bf16*)(ws + off);          off += (size_t)Bb * Hh * 2;
    unsigned* sync = (unsigned*)(ws + off);    off += 8;

    // h0, h1, sync are contiguous: zero them (h0 must be 0; sync must be 0).
    int nz = (Bb * Hh * 2 * 2 + 8) / 4;
    zero_u32<<<(nz + 255) / 256, 256, 0, stream>>>((unsigned*)h0, nz);

    pack_wmma_b<<<((Ii/32)*(Hh/16)*512 + 255) / 256, 256, 0, stream>>>(W_ih, wih_pk, Ii, Hh);
    pack_wmma_b<<<((Hh/32)*(Hh/16)*512 + 255) / 256, 256, 0, stream>>>(W_hh, whh_pk, Hh, Hh);
    pack_wmma_b<<<((Hh/32)*(Oo/16)*512 + 255) / 256, 256, 0, stream>>>(W_fc, wfc_pk, Hh, Oo);

    // 2048 row-tiles x 64 col-tiles = 131072 waves = 16384 blocks of 8 waves.
    xp_gemm<<<(Tt * Bb / 16) * (Hh / 16) / 8, 256, 0, stream>>>(x, wih_pk, b_ih, b_hh, xp);

    // Persistent scan: 128 co-resident blocks, split-K x4, 512 barriered steps.
    rnn_scan<<<128, 256, 0, stream>>>(whh_pk, xp, h0, h1, sync, 128);

    // Final h is in h0 after an even number of steps.
    head_gemm<<<8, 256, 0, stream>>>(h0, wfc_pk, b_fc, (float*)d_out);
}
